// CapsuleNet_17188459118750
// MI455X (gfx1250) — matrix operand order repaired and checked by reading, not compiled, not run on hardware
//
#include <hip/hip_runtime.h>
#include <hip/hip_bf16.h>
#include <stdint.h>

// ---------------------------------------------------------------------------
// Types
// ---------------------------------------------------------------------------
typedef __bf16 bf16;
typedef bf16  v16bf __attribute__((ext_vector_type(16)));
typedef float v8f   __attribute__((ext_vector_type(8)));
typedef unsigned int u32;
typedef u32 u32x4 __attribute__((ext_vector_type(4)));
typedef int i32x4 __attribute__((ext_vector_type(4)));
typedef int i32x8 __attribute__((ext_vector_type(8)));

// ---------------------------------------------------------------------------
// WMMA helper: D = A(16x32 bf16) * B(32x16 bf16) + C(16x16 f32)
// ---------------------------------------------------------------------------
__device__ __forceinline__ v8f wmma_bf16(v16bf a, v16bf b, v8f c) {
  return __builtin_amdgcn_wmma_f32_16x16x32_bf16(false, a, false, b,
                                                 (short)0, c, false, false);
}

// ---------------------------------------------------------------------------
// TDM: tile load Global -> LDS (one issue per wave; caller handles TENSORcnt)
// D# packed per CDNA5 ISA 8.3/8.4. dsz_code: 0=1B 1=2B 2=4B.
// Dims/tile/stride in data_size units.
// ---------------------------------------------------------------------------
__device__ __forceinline__ void tdm_load_2d(u32 lds_addr, const void* gptr,
                                            u32 dsz_code, u32 td0, u32 td1,
                                            u32 tile0, u32 tile1, u32 stride0) {
  unsigned long long ga = (unsigned long long)gptr;
  u32x4 g0;
  g0.x = 1u;                                   // count=1, user descriptor
  g0.y = lds_addr;                             // LDS byte address
  g0.z = (u32)ga;                              // global_addr[31:0]
  g0.w = (u32)((ga >> 32) & 0x1FFFFFFull) | (2u << 30);  // addr[56:32]|type=2
  i32x8 g1;
  g1[0] = (int)(dsz_code << 16);               // wg_mask=0, data_size
  g1[1] = (int)((td0 & 0xFFFFu) << 16);        // tensor_dim0[15:0]
  g1[2] = (int)((td0 >> 16) | ((td1 & 0xFFFFu) << 16));
  g1[3] = (int)((td1 >> 16) | (tile0 << 16));  // tile_dim0
  g1[4] = (int)(tile1 & 0xFFFFu);              // tile_dim1 (tile_dim2=0)
  g1[5] = (int)stride0;                        // tensor_dim0_stride[31:0]
  g1[6] = 0;
  g1[7] = 0;
  i32x4 g2 = {0, 0, 0, 0};
  i32x4 g3 = {0, 0, 0, 0};
#if __clang_major__ >= 23
  i32x8 g4 = {0, 0, 0, 0, 0, 0, 0, 0};
  __builtin_amdgcn_tensor_load_to_lds(g0, g1, g2, g3, g4, 0);
#else
  __builtin_amdgcn_tensor_load_to_lds(g0, g1, g2, g3, 0);
#endif
}

__device__ __forceinline__ u32 lds_off_u32(const void* p) {
  // Flat LDS addresses carry the LDS offset in the low 32 bits.
  return (u32)(unsigned long long)p;
}

// ---------------------------------------------------------------------------
// B-fragment packing. B operand (32x16 bf16) per wave, contiguous k order (A uses the same):
//   lane = (n&15) + 16*((k&31)>>4), element e = k&15.
// We store B as tiles of 512 bf16 so each lane loads one contiguous 32-byte
// fragment (2 x b128) instead of 16 strided 16-bit loads.
//   layout: Bp[(kt*Np + nt)*512 + lane*16 + e]
// ---------------------------------------------------------------------------
__global__ void k_pack_bfrag(const float* __restrict__ src, bf16* __restrict__ dst,
                             int Ksrc, int N, int Kp, int transposed) {
  int Np = N >> 4;
  int total = (Kp >> 5) * Np * 512;
  int i = blockIdx.x * 256 + threadIdx.x;
  if (i >= total) return;
  int kt  = i / (Np * 512);
  int rem = i - kt * (Np * 512);
  int nt  = rem / 512;
  int q   = rem - nt * 512;
  int lane = q >> 4, e = q & 15;
  int k = kt * 32 + ((lane >> 4) << 4) + e;
  int n = nt * 16 + (lane & 15);
  float v = 0.0f;
  if (k < Ksrc)
    v = transposed ? src[(size_t)n * Ksrc + k] : src[(size_t)k * N + n];
  dst[i] = (bf16)v;
}

// prim weights [256 out][256 in][81] -> fragment tiles, N-tile outermost so a
// (ntile, K-chunk) TDM transfer is a single contiguous region:
//   layout: B2p[nt][kt][512], kt = 0..767 (K padded to 256*96 = 24576)
__global__ void k_pack_b2frag(const float* __restrict__ w, bf16* __restrict__ dst) {
  int i = blockIdx.x * 256 + threadIdx.x;
  if (i >= 16 * 768 * 512) return;
  int nt  = i / (768 * 512);
  int rem = i - nt * (768 * 512);
  int kt  = rem / 512;
  int q   = rem - kt * 512;
  int lane = q >> 4, e = q & 15;
  int k = kt * 32 + ((lane >> 4) << 4) + e;       // 0..24575
  int n = nt * 16 + (lane & 15);
  int c = k / 96, j = k - c * 96;                 // 96 = padded taps/channel
  dst[i] = (j < 81) ? (bf16)w[((size_t)n * 256 + c) * 81 + j] : (bf16)0.0f;
}

// route weights [10][1152][8][16] f32 -> per-(c,n) single K-tile fragment
// (K = 8 zero-padded to 32): rp[(c*1152+n)*512 + lane*16 + e]
__global__ void k_pack_routefrag(const float* __restrict__ w, bf16* __restrict__ dst) {
  int i = blockIdx.x * 256 + threadIdx.x;
  if (i >= 10 * 1152 * 512) return;
  int cn = i / 512;                // c*1152 + n
  int q  = i - cn * 512;
  int lane = q >> 4, e = q & 15;
  int k = ((lane >> 4) << 4) + e;  // 0..31
  int o = lane & 15;
  dst[i] = (k < 8) ? (bf16)w[((size_t)cn * 8 + k) * 16 + o] : (bf16)0.0f;
}

// x [512][1][28][28] f32 -> A1 [204800][96] bf16 (im2col for conv1)
__global__ void k_im2col1(const float* __restrict__ x, bf16* __restrict__ A1) {
  int m = blockIdx.x * 256 + threadIdx.x;
  if (m >= 512 * 400) return;
  int b = m / 400, s = m % 400, oy = s / 20, ox = s % 20;
  const float* xp = x + (size_t)b * 784 + oy * 28 + ox;
  bf16* ap = A1 + (size_t)m * 96;
  for (int k = 0; k < 96; ++k) {
    float v = 0.0f;
    if (k < 81) { int ky = k / 9, kx = k % 9; v = xp[ky * 28 + kx]; }
    ap[k] = (bf16)v;
  }
}

// ---------------------------------------------------------------------------
// Generic WMMA GEMM: C[MxN] = act(A[MxK]bf16 * Bfrag bf16 + bias[N])
// block = 256 thr = 8 waves; wave w -> rows (blockIdx.x*8+w)*16,
// cols blockIdx.y*16. outMode 0: row-major [M][N]; outMode 1: m=(b,s) with
// spatial S -> out[b][n][s] (NCHW epilogue for conv1).
// ---------------------------------------------------------------------------
__global__ void k_gemm(const bf16* __restrict__ A, const bf16* __restrict__ Bp,
                       const float* __restrict__ bias,
                       float* __restrict__ Cf, bf16* __restrict__ Cb,
                       int M, int N, int K, int act, int outMode, int S) {
  int lane = threadIdx.x & 31, wave = threadIdx.x >> 5;
  int m0 = (blockIdx.x * 8 + wave) * 16;
  int n0 = blockIdx.y * 16;
  if (m0 >= M) return;
  int hf = lane >> 4, r = lane & 15;
  int Np = N >> 4;
  v8f acc = {0.f, 0.f, 0.f, 0.f, 0.f, 0.f, 0.f, 0.f};
#pragma unroll 1
  for (int k0 = 0; k0 < K; k0 += 32) {
    v16bf a;
    // A frag: lane row = m0+r; K(e) = e + 16*hf
    const bf16* ap = A + (size_t)(m0 + r) * K + k0 + 16 * hf;
#pragma unroll
    for (int e = 0; e < 16; ++e) a[e] = ap[e];
    // B frag: one contiguous 32B vector per lane
    v16bf b = *(const v16bf*)(Bp + ((size_t)(k0 >> 5) * Np + (n0 >> 4)) * 512 +
                              (size_t)lane * 16);
    acc = wmma_bf16(a, b, acc);
  }
#pragma unroll
  for (int g = 0; g < 8; ++g) {
    int mm = m0 + g + 8 * hf;   // D: row = vgpr + 8*half
    int nn = n0 + r;            //    col = lane&15
    float v = acc[g] + (bias ? bias[nn] : 0.0f);
    if (act == 1) v = fmaxf(v, 0.0f);
    else if (act == 2) v = 1.0f / (1.0f + __expf(-v));
    if (outMode == 0) {
      if (Cf) Cf[(size_t)mm * N + nn] = v;
      if (Cb) Cb[(size_t)mm * N + nn] = (bf16)v;
    } else {
      int bb = mm / S, ss = mm % S;
      size_t o = ((size_t)bb * N + nn) * S + ss;
      if (Cf) Cf[o] = v;
      if (Cb) Cb[o] = (bf16)v;
    }
  }
}

// ---------------------------------------------------------------------------
// Primary-capsule conv as implicit-im2col WMMA GEMM.
// M=18432 (b, 6x6 pos), N=256, K=24576 (256 ch * 96 padded taps).
// Fragment-major B panels streamed Global->LDS by the Tensor Data Mover with
// double buffering: issue chunk i+1, s_wait_tensorcnt(1) -> chunk i resident,
// compute overlapped with the DMA. A gathered on the fly from h (bf16 NCHW).
// ---------------------------------------------------------------------------
#define KC 1024                       // K elements per chunk = 32 k-tiles
#define NCHUNK (24576 / KC)           // 24
__global__ void k_prim_gemm(const bf16* __restrict__ h, const bf16* __restrict__ B2p,
                            const float* __restrict__ bias, float* __restrict__ pout) {
  __shared__ __attribute__((aligned(32))) bf16 sB[2][KC * 16];   // 2 x 32 KB
  int lane = threadIdx.x & 31, wave = threadIdx.x >> 5;
  int m0 = (blockIdx.x * 8 + wave) * 16;
  int nt = blockIdx.y;                       // N-tile
  int hf = lane >> 4, r = lane & 15;
  int row = m0 + r;
  int b = row / 36, s = row % 36;
  int iy0 = (s / 6) * 2, ix0 = (s % 6) * 2;
  const bf16* hb = h + (size_t)b * (256 * 400) + iy0 * 20 + ix0;
  v8f acc = {0.f, 0.f, 0.f, 0.f, 0.f, 0.f, 0.f, 0.f};
  u32 lds0 = lds_off_u32(&sB[0][0]);
  u32 lds1 = lds_off_u32(&sB[1][0]);
  const bf16* bsrc = B2p + (size_t)nt * 768 * 512;   // this block's N-tile

  if (threadIdx.x < 32)    // prologue: chunk 0 -> buf 0
    tdm_load_2d(lds0, bsrc, 1, 16384, 1, 16384, 0, 16384);

#pragma unroll 1
  for (int chunk = 0; chunk < NCHUNK; ++chunk) {
    int cur = chunk & 1;
    if (chunk + 1 < NCHUNK) {
      __syncthreads();               // everyone done reading buf[cur^1]
      if (threadIdx.x < 32) {
        tdm_load_2d(cur ? lds0 : lds1,
                    bsrc + (size_t)(chunk + 1) * 32 * 512,
                    1, 16384, 1, 16384, 0, 16384);
        __builtin_amdgcn_s_wait_tensorcnt(1);   // chunk `chunk` resident
      }
    } else {
      if (threadIdx.x < 32) __builtin_amdgcn_s_wait_tensorcnt(0);
    }
    __syncthreads();
    {   // prefetch upcoming activation rows (speculative, clamped)
      int pc = (chunk * KC) / 96 + 11; if (pc > 255) pc = 255;
      __builtin_prefetch(hb + (size_t)pc * 400, 0, 1);
    }
    const bf16* sbuf = &sB[cur][0];
    int kc0 = chunk * KC;
#pragma unroll 2
    for (int k = 0; k < KC; k += 32) {
      int kk0 = kc0 + k;
      int c = kk0 / 96, jb = kk0 - c * 96;     // 32-chunks never cross c
      const bf16* hc = hb + (size_t)c * 400;
      v16bf a;
#pragma unroll
      for (int e = 0; e < 16; ++e) {
        int j = jb + e + 16 * hf;
        bf16 v = (bf16)0.0f;
        if (j < 81) { int ky = j / 9, kx = j - ky * 9; v = hc[ky * 20 + kx]; }
        a[e] = v;
      }
      v16bf bm = *(const v16bf*)(sbuf + (size_t)(k >> 5) * 512 + (size_t)lane * 16);
      acc = wmma_bf16(a, bm, acc);
    }
  }
#pragma unroll
  for (int g = 0; g < 8; ++g) {
    int mm = m0 + g + 8 * hf;
    int bb = mm / 36, ss = mm % 36;
    int nn = nt * 16 + r;
    pout[((size_t)bb * 256 + nn) * 36 + ss] = acc[g] + bias[nn];
  }
}

// ---------------------------------------------------------------------------
// Squash primary capsules: p [512][256][36] f32 -> u [512][1152][8] bf16
// ---------------------------------------------------------------------------
__global__ void k_squash_prim(const float* __restrict__ p, bf16* __restrict__ u) {
  int idx = blockIdx.x * 256 + threadIdx.x;
  if (idx >= 512 * 1152) return;
  int b = idx / 1152, n = idx % 1152;
  int cc = n / 36, s = n % 36;
  float t[8]; float sn = 0.0f;
#pragma unroll
  for (int i = 0; i < 8; ++i) {
    float v = p[((size_t)b * 256 + (i * 32 + cc)) * 36 + s];
    t[i] = v; sn += v * v;
  }
  float scale = (sn / (1.0f + sn)) * rsqrtf(sn);
  bf16* up = u + (size_t)idx * 8;
#pragma unroll
  for (int i = 0; i < 8; ++i) up[i] = (bf16)(t[i] * scale);
}

// ---------------------------------------------------------------------------
// Priors: per (c,n): [512 x 8] @ [8 x 16] -> priors bf16 [c][b][n][16].
// One WMMA per wave, K padded 8 -> 32 with zeros (B pre-padded in fragment).
// ---------------------------------------------------------------------------
__global__ void k_priors(const bf16* __restrict__ u, const bf16* __restrict__ rp,
                         bf16* __restrict__ priors) {
  int n = blockIdx.x;          // route node
  int c = blockIdx.y;          // class
  int lane = threadIdx.x & 31, wave = threadIdx.x >> 5;
  int m0 = (blockIdx.z * 8 + wave) * 16;   // batch base
  int hf = lane >> 4, r = lane & 15;
  v16bf a;
  const bf16* up = u + ((size_t)(m0 + r) * 1152 + n) * 8;
#pragma unroll
  for (int e = 0; e < 16; ++e) {
    int kk = e + 16 * hf;
    a[e] = (kk < 8) ? up[kk] : (bf16)0.0f;
  }
  v16bf b = *(const v16bf*)(rp + ((size_t)c * 1152 + n) * 512 + (size_t)lane * 16);
  v8f acc = {0.f, 0.f, 0.f, 0.f, 0.f, 0.f, 0.f, 0.f};
  acc = wmma_bf16(a, b, acc);
#pragma unroll
  for (int g = 0; g < 8; ++g) {
    int bb = m0 + g + 8 * hf;
    priors[(((size_t)c * 512 + bb) * 1152 + n) * 16 + r] = (bf16)acc[g];
  }
}

// ---------------------------------------------------------------------------
// Dynamic routing, fully LDS-resident. One block per (b,c).
// Prior tile (1152x16 bf16 = 36 KB) TDM-loaded into LDS once; 3 iterations
// of softmax / weighted-sum / squash / logit update run from LDS with
// LDS float atomics for the block reductions.
// ---------------------------------------------------------------------------
__global__ void k_routing(const bf16* __restrict__ priors,
                          float* __restrict__ v_out, float* __restrict__ cls_out) {
  __shared__ __attribute__((aligned(32))) bf16 sp[1152 * 16];
  __shared__ float slog[1152];
  __shared__ float sprob[1152];
  __shared__ float s_out[16];
  __shared__ float s_vec[16];
  __shared__ float s_den;
  __shared__ float s_sn;
  int b = blockIdx.x, c = blockIdx.y;
  int t = threadIdx.x;
  if (t < 32) {
    tdm_load_2d(lds_off_u32(&sp[0]),
                priors + ((size_t)c * 512 + b) * (1152 * 16),
                1, 18432, 1, 18432, 0, 18432);
  }
  for (int n = t; n < 1152; n += 256) slog[n] = 0.0f;
  if (t < 16) s_out[t] = 0.0f;
  if (t == 0) s_den = 0.0f;
  if (t < 32) __builtin_amdgcn_s_wait_tensorcnt(0);
  __syncthreads();

  for (int it = 0; it < 3; ++it) {
    float dpart = 0.0f;
    for (int n = t; n < 1152; n += 256) {
      float e = __expf(slog[n]);
      sprob[n] = e;
      dpart += e;
    }
    atomicAdd(&s_den, dpart);
    __syncthreads();
    float inv = 1.0f / s_den;
    float accL[16];
#pragma unroll
    for (int o = 0; o < 16; ++o) accL[o] = 0.0f;
    for (int n = t; n < 1152; n += 256) {
      float pr = sprob[n] * inv;
#pragma unroll
      for (int o = 0; o < 16; ++o) accL[o] += pr * (float)sp[n * 16 + o];
    }
#pragma unroll
    for (int o = 0; o < 16; ++o) atomicAdd(&s_out[o], accL[o]);
    __syncthreads();
    if (t == 0) {
      float sn = 0.0f;
      for (int o = 0; o < 16; ++o) sn += s_out[o] * s_out[o];
      float scale = (sn / (1.0f + sn)) * rsqrtf(sn);
      for (int o = 0; o < 16; ++o) s_vec[o] = s_out[o] * scale;
      s_sn = sn;
      s_den = 0.0f;
      for (int o = 0; o < 16; ++o) s_out[o] = 0.0f;
    }
    __syncthreads();
    if (it < 2) {
      for (int n = t; n < 1152; n += 256) {
        float d = 0.0f;
#pragma unroll
        for (int o = 0; o < 16; ++o) d += (float)sp[n * 16 + o] * s_vec[o];
        slog[n] += d;
      }
      __syncthreads();
    }
  }
  if (t < 16) v_out[((size_t)b * 10 + c) * 16 + t] = s_vec[t];
  if (t == 0) cls_out[b * 10 + c] = s_sn / (1.0f + s_sn);   // ||v||
}

// ---------------------------------------------------------------------------
// Argmax over classes, one-hot y_pred, build masked decoder input (bf16).
// ---------------------------------------------------------------------------
__global__ void k_mask(const float* __restrict__ v, const float* __restrict__ cls,
                       float* __restrict__ y_pred, bf16* __restrict__ masked) {
  int b = blockIdx.x * 64 + threadIdx.x;
  if (b >= 512) return;
  int best = 0; float bv = cls[b * 10];
  for (int c = 1; c < 10; ++c) {
    float cv = cls[b * 10 + c];
    if (cv > bv) { bv = cv; best = c; }
  }
  for (int c = 0; c < 10; ++c) {
    y_pred[b * 10 + c] = (c == best) ? 1.0f : 0.0f;
    for (int o = 0; o < 16; ++o)
      masked[b * 160 + c * 16 + o] =
          (c == best) ? (bf16)v[((size_t)b * 10 + c) * 16 + o] : (bf16)0.0f;
  }
}

// ---------------------------------------------------------------------------
// Launch
// ---------------------------------------------------------------------------
extern "C" void kernel_launch(void* const* d_in, const int* in_sizes, int n_in,
                              void* d_out, int out_size, void* d_ws, size_t ws_size,
                              hipStream_t stream) {
  (void)in_sizes; (void)n_in; (void)out_size; (void)ws_size;
  const float* x       = (const float*)d_in[0];
  const float* conv1_w = (const float*)d_in[1];
  const float* conv1_b = (const float*)d_in[2];
  const float* prim_w  = (const float*)d_in[3];
  const float* prim_b  = (const float*)d_in[4];
  const float* route_w = (const float*)d_in[5];
  const float* dec_w1  = (const float*)d_in[6];
  const float* dec_b1  = (const float*)d_in[7];
  const float* dec_w2  = (const float*)d_in[8];
  const float* dec_b2  = (const float*)d_in[9];
  const float* dec_w3  = (const float*)d_in[10];
  const float* dec_b3  = (const float*)d_in[11];

  float* out = (float*)d_out;
  float* out_ypred = out;                        // [512][10]
  float* out_recon = out + 512 * 10;             // [512][784]
  float* out_cls   = out + 512 * 10 + 512 * 784; // [512][10]

  char* ws = (char*)d_ws;
  size_t off = 0;
  auto walloc = [&](size_t bytes) -> void* {
    void* p = (void*)(ws + off);
    off += (bytes + 255) & ~(size_t)255;
    return p;
  };
  bf16*  B1p    = (bf16*)walloc((size_t)3 * 16 * 512 * 2);        // conv1 B frags
  bf16*  A1     = (bf16*)walloc((size_t)204800 * 96 * 2);
  bf16*  h      = (bf16*)walloc((size_t)512 * 256 * 400 * 2);
  bf16*  B2p    = (bf16*)walloc((size_t)16 * 768 * 512 * 2);      // prim B frags
  float* pbuf   = (float*)walloc((size_t)512 * 256 * 36 * 4);
  bf16*  u      = (bf16*)walloc((size_t)512 * 1152 * 8 * 2);
  bf16*  rp     = (bf16*)walloc((size_t)10 * 1152 * 512 * 2);     // route frags
  bf16*  priors = (bf16*)walloc((size_t)10 * 512 * 1152 * 16 * 2);
  float* v      = (float*)walloc((size_t)512 * 10 * 16 * 4);
  bf16*  masked = (bf16*)walloc((size_t)512 * 160 * 2);
  bf16*  w1p    = (bf16*)walloc((size_t)5 * 32 * 512 * 2);
  bf16*  w2p    = (bf16*)walloc((size_t)16 * 64 * 512 * 2);
  bf16*  w3p    = (bf16*)walloc((size_t)32 * 49 * 512 * 2);
  bf16*  d1b    = (bf16*)walloc((size_t)512 * 512 * 2);
  bf16*  d2b    = (bf16*)walloc((size_t)512 * 1024 * 2);

  // Pack weights into WMMA-fragment layout; im2col conv1 input
  k_pack_bfrag<<<dim3(96), 256, 0, stream>>>(conv1_w, B1p, 81, 256, 96, 1);
  k_im2col1<<<dim3(800), 256, 0, stream>>>(x, A1);
  k_pack_b2frag<<<dim3(24576), 256, 0, stream>>>(prim_w, B2p);
  k_pack_routefrag<<<dim3(23040), 256, 0, stream>>>(route_w, rp);
  k_pack_bfrag<<<dim3(320), 256, 0, stream>>>(dec_w1, w1p, 160, 512, 160, 0);
  k_pack_bfrag<<<dim3(2048), 256, 0, stream>>>(dec_w2, w2p, 512, 1024, 512, 0);
  k_pack_bfrag<<<dim3(3136), 256, 0, stream>>>(dec_w3, w3p, 1024, 784, 1024, 0);

  // conv1 (+bias+ReLU) as GEMM, epilogue writes bf16 NCHW h
  k_gemm<<<dim3(1600, 16), 256, 0, stream>>>(A1, B1p, conv1_b, nullptr, h,
                                             204800, 256, 96, /*relu*/1,
                                             /*NCHW*/1, 400);
  // primary caps: implicit-im2col WMMA GEMM, double-buffered TDM B panels
  k_prim_gemm<<<dim3(144, 16), 256, 0, stream>>>(h, B2p, prim_b, pbuf);
  k_squash_prim<<<dim3(2304), 256, 0, stream>>>(pbuf, u);
  k_priors<<<dim3(1152, 10, 4), 256, 0, stream>>>(u, rp, priors);
  k_routing<<<dim3(512, 10), 256, 0, stream>>>(priors, v, out_cls);
  k_mask<<<dim3(8), 64, 0, stream>>>(v, out_cls, out_ypred, masked);

  // decoder MLP
  k_gemm<<<dim3(4, 32), 256, 0, stream>>>(masked, w1p, dec_b1, nullptr, d1b,
                                          512, 512, 160, 1, 0, 0);
  k_gemm<<<dim3(4, 64), 256, 0, stream>>>(d1b, w2p, dec_b2, nullptr, d2b,
                                          512, 1024, 512, 1, 0, 0);
  k_gemm<<<dim3(4, 49), 256, 0, stream>>>(d2b, w3p, dec_b3, out_recon, nullptr,
                                          512, 784, 1024, /*sigmoid*/2, 0, 0);
}
